// LargeWindowAttention_38912403701870
// MI455X (gfx1250) — matrix operand order repaired and checked by reading, compiled, not run on hardware
//
#include <hip/hip_runtime.h>
#include <hip/hip_bf16.h>
#include <stdint.h>

// ---------------------------------------------------------------------------
// CDNA5 / gfx1250 wave32 WMMA implementation of LargeWindowAttention.
// Dense math: v_wmma_f32_16x16x32_bf16. Shared operands staged in LDS via
// global_load_async_to_lds_b128 (ASYNCcnt) to kill redundant L2 traffic.
// ---------------------------------------------------------------------------

typedef __bf16 bf16_t;
typedef __attribute__((ext_vector_type(16))) __bf16 v16bf;
typedef __attribute__((ext_vector_type(8)))  __bf16 v8bf;
typedef __attribute__((ext_vector_type(8)))  float  v8f;

#define C_DIM 512
#define HWDIM 128
#define LTOK  64
#define NWIN  256
#define NHEAD 8
#define HD    64
#define MROWS 16384   // NWIN * LTOK

// ---- async copy global -> LDS (per-lane), ASYNCcnt tracked ----------------
__device__ __forceinline__ void async_copy_b128(uint32_t lds_off, const void* gptr) {
  asm volatile("global_load_async_to_lds_b128 %0, %1, off"
               :: "v"(lds_off), "v"((uint64_t)(uintptr_t)gptr)
               : "memory");
}
#define WAIT_ASYNC(n) asm volatile("s_wait_asynccnt " #n ::: "memory")
#define WAIT_DS0()    asm volatile("s_wait_dscnt 0" ::: "memory")

// ---- WMMA fragment helpers (layouts per CDNA5 ISA §7.12.2) -----

// A fragment 16x32 bf16: lane r (=lane&15) holds row r; lane group g (=lane>>4)
// selects K-chunks {g*8 .. g*8+7} and {16+g*8 .. 16+g*8+7}.
__device__ __forceinline__ v16bf load_a_frag(const bf16_t* base, int rowStride, int lane) {
  const int r = lane & 15, g = lane >> 4;
  const bf16_t* p = base + r * rowStride + g * 8;
  union { v16bf v; v8bf h[2]; } u;
  u.h[0] = *(const v8bf*)(p);
  u.h[1] = *(const v8bf*)(p + 16);
  return u.v;
}

// B fragment 32x16 bf16: lane n (=lane&15) holds column n; lane group g selects
// K-chunks {g*8 .. g*8+7} and {16+g*8 .. 16+g*8+7}. colStride = elements between columns.
__device__ __forceinline__ v16bf load_b_frag(const bf16_t* base, int colStride, int lane) {
  const int n = lane & 15, g = lane >> 4;
  const bf16_t* p = base + n * colStride + g * 8;
  union { v16bf v; v8bf h[2]; } u;
  u.h[0] = *(const v8bf*)(p);
  u.h[1] = *(const v8bf*)(p + 16);
  return u.v;
}

__device__ __forceinline__ v8f wmma_bf16(v16bf a, v16bf b, v8f c) {
  return __builtin_amdgcn_wmma_f32_16x16x32_bf16(false, a, false, b, (short)0, c, false, false);
}

// ---------------------------------------------------------------------------
// K0: fp32 -> bf16 weight conversion
// ---------------------------------------------------------------------------
__global__ void cvt_bf16_kernel(const float* __restrict__ src, bf16_t* __restrict__ dst, int n) {
  int i = blockIdx.x * 256 + threadIdx.x;
  if (i < n) dst[i] = (bf16_t)src[i];
}

// ---------------------------------------------------------------------------
// K1: gather non-overlapping 8x8 windows -> qb [MROWS][C_DIM] bf16
// ---------------------------------------------------------------------------
__global__ void build_q_kernel(const float* __restrict__ x, bf16_t* __restrict__ qb) {
  int idx = blockIdx.x * 256 + threadIdx.x;          // < MROWS*C_DIM
  int c = idx & (C_DIM - 1);
  int t = idx >> 9;                                  // b*64 + l
  int l = t & 63, b = t >> 6;
  int wy = b >> 4, wx = b & 15;
  int py = l >> 3, px = l & 7;
  int y = wy * 8 + py, xx = wx * 8 + px;
  qb[idx] = (bf16_t)x[c * (HWDIM * HWDIM) + y * HWDIM + xx];
}

// ---------------------------------------------------------------------------
// K2: overlapped-window 2x2 average pool + LayerNorm over C.
// Output tb bf16 layout [B][c][l]  (l contiguous).
// ---------------------------------------------------------------------------
__global__ __launch_bounds__(128) void build_ctx_kernel(
    const float* __restrict__ x, const float* __restrict__ ln_w,
    const float* __restrict__ ln_b, bf16_t* __restrict__ tb) {
  __shared__ float rs[128], rq[128];
  int bl = blockIdx.x;                   // b*64 + l
  int l = bl & 63, b = bl >> 6;
  int wy = b >> 4, wx = b & 15;
  int py = l >> 3, px = l & 7;
  int y0 = wy * 8 - 4 + 2 * py;          // pad = (16-8)/2 = 4
  int x0 = wx * 8 - 4 + 2 * px;
  int tid = threadIdx.x;

  float pooled[4];
  float s1 = 0.f, s2 = 0.f;
#pragma unroll
  for (int j = 0; j < 4; ++j) {
    int c = tid + j * 128;
    const float* xc = x + c * (HWDIM * HWDIM);
    float acc = 0.f;
#pragma unroll
    for (int dy = 0; dy < 2; ++dy) {
      int y = y0 + dy;
      if ((unsigned)y < HWDIM) {
#pragma unroll
        for (int dx = 0; dx < 2; ++dx) {
          int xq = x0 + dx;
          if ((unsigned)xq < HWDIM) acc += xc[y * HWDIM + xq];
        }
      }
    }
    float p = acc * 0.25f;
    pooled[j] = p;
    s1 += p;
    s2 += p * p;
  }
  rs[tid] = s1; rq[tid] = s2;
  __syncthreads();
  for (int off = 64; off > 0; off >>= 1) {
    if (tid < off) { rs[tid] += rs[tid + off]; rq[tid] += rq[tid + off]; }
    __syncthreads();
  }
  float mean = rs[0] * (1.f / 512.f);
  float var  = rq[0] * (1.f / 512.f) - mean * mean;
  float rstd = rsqrtf(var + 1e-6f);
#pragma unroll
  for (int j = 0; j < 4; ++j) {
    int c = tid + j * 128;
    float v = (pooled[j] - mean) * rstd * ln_w[c] + ln_b[c];
    tb[(b * C_DIM + c) * LTOK + l] = (bf16_t)v;
  }
}

// ---------------------------------------------------------------------------
// K3: token mix. Per (b,h): mixed^T(m,cc) = mix_w[h](m,:) . t(cc,:)  via WMMA,
// + mix_b[h][m] + residual t(cc,m).  Output ctxb bf16 [B][l][c].
// ---------------------------------------------------------------------------
__global__ __launch_bounds__(128) void mix_kernel(
    const bf16_t* __restrict__ tb, const bf16_t* __restrict__ mixw,
    const float* __restrict__ mix_b, bf16_t* __restrict__ ctxb) {
  int bh = blockIdx.x;                   // b*8 + h
  int b = bh >> 3, h = bh & 7;
  int wave = threadIdx.x >> 5;           // 0..3 -> m-tile
  int lane = threadIdx.x & 31;
  int g = lane >> 4;
  int m0 = wave * 16;

  v8f acc[4] = {};
  const bf16_t* Abase = mixw + h * 4096 + m0 * 64;        // (m x l), ld=64
  const bf16_t* Bcol0 = tb + (b * C_DIM + h * HD) * LTOK; // column cc over l
#pragma unroll
  for (int k0 = 0; k0 < 64; k0 += 32) {
    v16bf a = load_a_frag(Abase + k0, 64, lane);
#pragma unroll
    for (int j = 0; j < 4; ++j) {
      v16bf bb = load_b_frag(Bcol0 + (j * 16) * LTOK + k0, LTOK, lane);
      acc[j] = wmma_bf16(a, bb, acc[j]);
    }
  }
#pragma unroll
  for (int j = 0; j < 4; ++j) {
    int cc = j * 16 + (lane & 15);
#pragma unroll
    for (int i = 0; i < 8; ++i) {
      int m = m0 + i + 8 * g;            // C/D layout: vgpr i -> rows i / i+8
      float resid = (float)tb[(b * C_DIM + h * HD + cc) * LTOK + m];
      float val = acc[j][i] + mix_b[h * HD + m] + resid;
      ctxb[(b * LTOK + m) * C_DIM + h * HD + cc] = (bf16_t)val;
    }
  }
}

// ---------------------------------------------------------------------------
// K4: GEMM Out[M][N] = A[M][K] * W[N][K]^T + bias, bf16 in/out.
// Block = 8 waves, block tile 256x64, wave tile 32x64, k-step 64.
// B tile (shared by all waves) double-buffer staged in LDS via async loads
// (2 x b128 per thread per stage); 16 WMMAs between barrier pairs.
// ---------------------------------------------------------------------------
__global__ __launch_bounds__(256) void gemm_bf16_kernel(
    const bf16_t* __restrict__ A, const bf16_t* __restrict__ W,
    const float* __restrict__ bias, bf16_t* __restrict__ Out,
    int M, int N, int K) {
  __shared__ __align__(16) bf16_t Btile[2][64 * 64];   // 2 x 8 KB
  const int tid  = threadIdx.x;
  const int lane = tid & 31;
  const int wave = tid >> 5;
  const int g = lane >> 4;
  const int n0 = blockIdx.x * 64;
  const int m0 = blockIdx.y * 256 + wave * 32;

  const int brow = tid >> 2;           // 0..63 : B column (weight row)
  const int bseg = (tid & 3) * 16;     // element offset in 64-wide k-slab
  const bf16_t* bsrc = W + (n0 + brow) * K + bseg;
  const uint32_t l0 = (uint32_t)(uintptr_t)&Btile[0][brow * 64 + bseg];
  const uint32_t l1 = (uint32_t)(uintptr_t)&Btile[1][brow * 64 + bseg];

  async_copy_b128(l0, bsrc);           // prologue: stage k0 = 0
  async_copy_b128(l0 + 16, bsrc + 8);

  v8f acc[2][4] = {};
  const bf16_t* Abase = A + m0 * K;
  for (int k0 = 0; k0 < K; k0 += 64) {
    const int cur = (k0 >> 6) & 1;
    if (k0 + 64 < K) {
      uint32_t nl = cur ? l0 : l1;
      async_copy_b128(nl, bsrc + k0 + 64);
      async_copy_b128(nl + 16, bsrc + k0 + 64 + 8);
      WAIT_ASYNC(2);                   // current tile landed; next may fly
    } else {
      WAIT_ASYNC(0);
    }
    __syncthreads();
#pragma unroll
    for (int ks = 0; ks < 2; ++ks) {
      v16bf a0 = load_a_frag(Abase + k0 + ks * 32, K, lane);
      v16bf a1 = load_a_frag(Abase + 16 * K + k0 + ks * 32, K, lane);
#pragma unroll
      for (int j = 0; j < 4; ++j) {
        v16bf b = load_b_frag(&Btile[cur][(j * 16) * 64 + ks * 32], 64, lane);
        acc[0][j] = wmma_bf16(a0, b, acc[0][j]);
        acc[1][j] = wmma_bf16(a1, b, acc[1][j]);
      }
    }
    __syncthreads();                   // done reading before overwrite
  }
#pragma unroll
  for (int r = 0; r < 2; ++r) {
#pragma unroll
    for (int j = 0; j < 4; ++j) {
      int col = n0 + j * 16 + (lane & 15);
      float bv = bias ? bias[col] : 0.f;
#pragma unroll
      for (int i = 0; i < 8; ++i) {
        int row = m0 + r * 16 + i + 8 * g;
        Out[row * N + col] = (bf16_t)(acc[r][j][i] + bv);
      }
    }
  }
}

// ---------------------------------------------------------------------------
// K5: V transpose -> Vt [l][h][d][t]  (t contiguous: P.V B operand)
// ---------------------------------------------------------------------------
__global__ void transpose_v_kernel(const bf16_t* __restrict__ Vb, bf16_t* __restrict__ Vt) {
  int idx = blockIdx.x * 256 + threadIdx.x;   // 8388608
  int t = idx & 255;
  int r = idx >> 8;                           // (l*8+h)*64 + d
  int d = r & 63;
  int lh = r >> 6;
  int h = lh & 7, l = lh >> 3;
  Vt[idx] = Vb[(t * LTOK + l) * C_DIM + h * HD + d];
}

// ---------------------------------------------------------------------------
// K6: attention. grid = l(64)*h(8)*part(2); 8 waves/block; wave owns 16 query
// windows of one (l,h) problem. The 256x64 K slab and 64x256 V^T slab (shared
// by all waves) are async-staged into one recycled 32 KB LDS tile. S (16x256)
// lives in VGPRs; softmax via width-16 shuffles; P goes through per-wave LDS
// to form A fragments for the P.V WMMAs.
// ---------------------------------------------------------------------------
__global__ __launch_bounds__(256) void attn_kernel(
    const bf16_t* __restrict__ Qb, const bf16_t* __restrict__ Kb,
    const bf16_t* __restrict__ Vt, bf16_t* __restrict__ Ob) {
  __shared__ __align__(16) bf16_t tile[256 * 64];   // 32 KB: K slab, then V^T
  __shared__ __align__(16) bf16_t pbuf[8][16 * 32]; // 8 KB per-wave P staging
  int bid = blockIdx.x;
  int part = bid & 1;
  int h = (bid >> 1) & 7;
  int l = bid >> 4;
  int tid = threadIdx.x;
  int wave = tid >> 5;
  int lane = tid & 31;
  int g = lane >> 4;
  int s0 = (part * 8 + wave) * 16;

  const int RS = LTOK * C_DIM;                // stride between window rows
  const bf16_t* Qbase = Qb + (s0 * LTOK + l) * C_DIM + h * HD;

  // ---- stage K slab: thread tid copies row t=tid (64 bf16 = 128 B) --------
  {
    const bf16_t* src = Kb + (tid * LTOK + l) * C_DIM + h * HD;
    uint32_t dst = (uint32_t)(uintptr_t)&tile[tid * 64];
#pragma unroll
    for (int q = 0; q < 8; ++q) async_copy_b128(dst + q * 16, src + q * 8);
  }
  WAIT_ASYNC(0);
  __syncthreads();

  // ---- scores S[s0..s0+15][0..255] = Q . K^T ------------------------------
  v8f S[16] = {};
#pragma unroll
  for (int kk = 0; kk < 64; kk += 32) {
    v16bf a = load_a_frag(Qbase + kk, RS, lane);
#pragma unroll
    for (int tt = 0; tt < 16; ++tt) {
      v16bf b = load_b_frag(&tile[(tt * 16) * 64 + kk], 64, lane);
      S[tt] = wmma_bf16(a, b, S[tt]);
    }
  }

  // ---- softmax over t (scale = 1/sqrt(64)); stats uniform per lane group
  // for row (i + 8*g), matching the C/D layout ------------------------------
  const float scale = 0.125f;
  float inv[8];
#pragma unroll
  for (int i = 0; i < 8; ++i) {
    float m = -3.0e38f;
#pragma unroll
    for (int tt = 0; tt < 16; ++tt) m = fmaxf(m, S[tt][i]);
#pragma unroll
    for (int off = 8; off > 0; off >>= 1) m = fmaxf(m, __shfl_xor(m, off, 16));
    float sum = 0.f;
#pragma unroll
    for (int tt = 0; tt < 16; ++tt) {
      float p = __expf(scale * (S[tt][i] - m));
      S[tt][i] = p;
      sum += p;
    }
#pragma unroll
    for (int off = 8; off > 0; off >>= 1) sum += __shfl_xor(sum, off, 16);
    inv[i] = 1.f / sum;
  }

  // ---- recycle tile for V^T: row d = tid/4, 64 bf16 per thread ------------
  __syncthreads();                             // everyone done with K slab
  {
    int d = tid >> 2;
    int seg = (tid & 3) * 64;                  // element offset in 256-wide row
    const bf16_t* src = Vt + ((l * NHEAD + h) * HD + d) * 256 + seg;
    uint32_t dst = (uint32_t)(uintptr_t)&tile[d * 256 + seg];
#pragma unroll
    for (int q = 0; q < 8; ++q) async_copy_b128(dst + q * 16, src + q * 8);
  }
  WAIT_ASYNC(0);
  __syncthreads();

  // ---- O = P . V^T : P staged through per-wave LDS for A fragments --------
  v8f Oacc[4] = {};
  bf16_t* pw = &pbuf[wave][0];
#pragma unroll
  for (int tc = 0; tc < 8; ++tc) {
#pragma unroll
    for (int half = 0; half < 2; ++half) {
      int tt = tc * 2 + half;
#pragma unroll
      for (int i = 0; i < 8; ++i) {
        int row = i + 8 * g;
        pw[row * 32 + half * 16 + (lane & 15)] = (bf16_t)S[tt][i];
      }
    }
    WAIT_DS0();                                // LDS store -> load ordering
    v16bf a = load_a_frag(pw, 32, lane);
#pragma unroll
    for (int j = 0; j < 4; ++j) {
      v16bf b = load_b_frag(&tile[(j * 16) * 256 + tc * 32], 256, lane);
      Oacc[j] = wmma_bf16(a, b, Oacc[j]);
    }
  }

#pragma unroll
  for (int j = 0; j < 4; ++j) {
    int d = j * 16 + (lane & 15);
#pragma unroll
    for (int i = 0; i < 8; ++i) {
      int srow = s0 + i + 8 * g;
      Ob[(srow * LTOK + l) * C_DIM + h * HD + d] = (bf16_t)(Oacc[j][i] * inv[i]);
    }
  }
}

// ---------------------------------------------------------------------------
// K7: output projection + bias + fused scatter back to NCHW fp32.
// Same tiling and async double-buffered B staging as the main GEMM.
// ---------------------------------------------------------------------------
__global__ __launch_bounds__(256) void out_proj_kernel(
    const bf16_t* __restrict__ A, const bf16_t* __restrict__ W,
    const float* __restrict__ bias, float* __restrict__ out) {
  __shared__ __align__(16) bf16_t Btile[2][64 * 64];
  const int tid  = threadIdx.x;
  const int lane = tid & 31;
  const int wave = tid >> 5;
  const int g = lane >> 4;
  const int n0 = blockIdx.x * 64;
  const int m0 = blockIdx.y * 256 + wave * 32;

  const int brow = tid >> 2;
  const int bseg = (tid & 3) * 16;
  const bf16_t* bsrc = W + (n0 + brow) * C_DIM + bseg;
  const uint32_t l0 = (uint32_t)(uintptr_t)&Btile[0][brow * 64 + bseg];
  const uint32_t l1 = (uint32_t)(uintptr_t)&Btile[1][brow * 64 + bseg];

  async_copy_b128(l0, bsrc);
  async_copy_b128(l0 + 16, bsrc + 8);

  v8f acc[2][4] = {};
  const bf16_t* Abase = A + m0 * C_DIM;
  for (int k0 = 0; k0 < C_DIM; k0 += 64) {
    const int cur = (k0 >> 6) & 1;
    if (k0 + 64 < C_DIM) {
      uint32_t nl = cur ? l0 : l1;
      async_copy_b128(nl, bsrc + k0 + 64);
      async_copy_b128(nl + 16, bsrc + k0 + 64 + 8);
      WAIT_ASYNC(2);
    } else {
      WAIT_ASYNC(0);
    }
    __syncthreads();
#pragma unroll
    for (int ks = 0; ks < 2; ++ks) {
      v16bf a0 = load_a_frag(Abase + k0 + ks * 32, C_DIM, lane);
      v16bf a1 = load_a_frag(Abase + 16 * C_DIM + k0 + ks * 32, C_DIM, lane);
#pragma unroll
      for (int j = 0; j < 4; ++j) {
        v16bf b = load_b_frag(&Btile[cur][(j * 16) * 64 + ks * 32], 64, lane);
        acc[0][j] = wmma_bf16(a0, b, acc[0][j]);
        acc[1][j] = wmma_bf16(a1, b, acc[1][j]);
      }
    }
    __syncthreads();
  }
#pragma unroll
  for (int r = 0; r < 2; ++r) {
#pragma unroll
    for (int j = 0; j < 4; ++j) {
      int c = n0 + j * 16 + (lane & 15);
      float bv = bias[c];
#pragma unroll
      for (int i = 0; i < 8; ++i) {
        int row = m0 + r * 16 + i + 8 * g;
        int s = row >> 6, l = row & 63;
        int wy = s >> 4, wx = s & 15;
        int py = l >> 3, px = l & 7;
        out[c * (HWDIM * HWDIM) + (wy * 8 + py) * HWDIM + (wx * 8 + px)] =
            acc[r][j][i] + bv;
      }
    }
  }
}

// ---------------------------------------------------------------------------
extern "C" void kernel_launch(void* const* d_in, const int* in_sizes, int n_in,
                              void* d_out, int out_size, void* d_ws, size_t ws_size,
                              hipStream_t stream) {
  const float* x     = (const float*)d_in[0];
  const float* ln_w  = (const float*)d_in[1];
  const float* ln_b  = (const float*)d_in[2];
  const float* mix_w = (const float*)d_in[3];
  const float* mix_b = (const float*)d_in[4];
  const float* in_w  = (const float*)d_in[5];
  const float* in_b  = (const float*)d_in[6];
  const float* out_w = (const float*)d_in[7];
  const float* out_b = (const float*)d_in[8];
  float* out = (float*)d_out;

  char* ws = (char*)d_ws;
  const size_t BIG = (size_t)MROWS * C_DIM * 2;   // 16,777,216 B per bf16 buffer
  bf16_t* wqkv = (bf16_t*)(ws);                   // 3*512*512 bf16
  bf16_t* wo   = (bf16_t*)(ws + 1572864);         // 512*512 bf16
  bf16_t* mixw = (bf16_t*)(ws + 2097152);         // 8*64*64 bf16
  char* big0   = ws + 2162688;
  bf16_t* qb   = (bf16_t*)(big0);                 // q activations  (later: Vt)
  bf16_t* tb   = (bf16_t*)(big0 + 1 * BIG);       // ctx LN, [B][c][l] (later: Ob)
  bf16_t* ctxb = (bf16_t*)(big0 + 2 * BIG);       // ctx post-mix, [B][l][c]
  bf16_t* Qb   = (bf16_t*)(big0 + 3 * BIG);
  bf16_t* Kb   = (bf16_t*)(big0 + 4 * BIG);
  bf16_t* Vb   = (bf16_t*)(big0 + 5 * BIG);
  bf16_t* Vt   = qb;   // qb dead after Q GEMM
  bf16_t* Ob   = tb;   // tb dead after mix

  cvt_bf16_kernel<<<(786432 + 255) / 256, 256, 0, stream>>>(in_w, wqkv, 786432);
  cvt_bf16_kernel<<<(262144 + 255) / 256, 256, 0, stream>>>(out_w, wo, 262144);
  cvt_bf16_kernel<<<(32768 + 255) / 256, 256, 0, stream>>>(mix_w, mixw, 32768);

  build_q_kernel<<<32768, 256, 0, stream>>>(x, qb);
  build_ctx_kernel<<<NWIN * LTOK, 128, 0, stream>>>(x, ln_w, ln_b, tb);
  mix_kernel<<<NWIN * NHEAD, 128, 0, stream>>>(tb, mixw, mix_b, ctxb);

  dim3 ggrid(C_DIM / 64, MROWS / 256);   // (8, 64)
  gemm_bf16_kernel<<<ggrid, 256, 0, stream>>>(qb,   wqkv,                 in_b,        Qb, MROWS, C_DIM, C_DIM);
  gemm_bf16_kernel<<<ggrid, 256, 0, stream>>>(ctxb, wqkv + 512 * 512,     in_b + 512,  Kb, MROWS, C_DIM, C_DIM);
  gemm_bf16_kernel<<<ggrid, 256, 0, stream>>>(ctxb, wqkv + 2 * 512 * 512, in_b + 1024, Vb, MROWS, C_DIM, C_DIM);

  transpose_v_kernel<<<32768, 256, 0, stream>>>(Vb, Vt);
  attn_kernel<<<LTOK * NHEAD * 2, 256, 0, stream>>>(Qb, Kb, Vt, Ob);
  out_proj_kernel<<<ggrid, 256, 0, stream>>>(Ob, wo, out_b, out);
}
